// VQProsodyEncoder_7258494730430
// MI455X (gfx1250) — matrix operand order repaired and checked by reading, not compiled, not run on hardware
//
#include <hip/hip_runtime.h>
#include <hip/hip_bf16.h>

// ---------------- problem constants ----------------
#define B_    16
#define T_    2048
#define H_    320
#define K_    5
#define NE_   2048
#define D_    256
#define TP_   (T_ + 4)               // time-padded rows (2 each side)
#define NFLAT_ (B_ * T_ * H_ / D_)   // 40960 flat VQ rows
#define NPB_   (T_ * H_ / D_)        // 2560 flat rows per batch (exact)
#define NTOT_  ((size_t)B_ * T_ * H_)// 10485760 elements
#define KDIM_  (H_ * K_)             // 1600 = GEMM K for conv
#define KBLK_  (KDIM_ / 32)          // 50 K-blocks of 32
#define NBLK_  (H_ / 16)             // 20 co-tiles of 16
#define WPER_  (KBLK_ * NBLK_ * 32 * 16) // 512000 packed u16 per layer

typedef __attribute__((ext_vector_type(16))) __bf16 v16bf;
typedef __attribute__((ext_vector_type(8)))  float  v8f;
typedef unsigned short u16;
typedef unsigned int   u32;

// ---------------- helpers ----------------
__device__ __forceinline__ u16 f2bf(float f) {        // RNE float->bf16
  u32 u = __float_as_uint(f);
  u32 r = u + 0x7FFFu + ((u >> 16) & 1u);
  return (u16)(r >> 16);
}

// build a 16-element bf16 operand from two contiguous 16-byte chunks
__device__ __forceinline__ v16bf load_bf16x16(const u16* p0, const u16* p1) {
  union { uint4 q[2]; v16bf v; } u;
  u.q[0] = *reinterpret_cast<const uint4*>(p0);
  u.q[1] = *reinterpret_cast<const uint4*>(p1);
  return u.v;
}

// async global->LDS copy of 16B (GVS mode: SGPR base + 32-bit VGPR offset).
// lds_off: LDS byte address (low 32 bits of the flat shared address).
__device__ __forceinline__ void async_load_b128(unsigned lds_off, const void* sbase,
                                                unsigned voff) {
  asm volatile("global_load_async_to_lds_b128 %0, %1, %2"
               :
               : "v"(lds_off), "v"(voff), "s"(sbase)
               : "memory");
}
__device__ __forceinline__ void wait_async0() {
  asm volatile("s_wait_asynccnt 0" ::: "memory");
}
__device__ __forceinline__ unsigned lds_addr_of(const void* p) {
  return (unsigned)(uintptr_t)p;    // flat LDS address: low 32 bits = LDS offset
}

__device__ __forceinline__ float blockReduceSum256(float v) {
  __shared__ float s[256];
  __syncthreads();
  s[threadIdx.x] = v;
  __syncthreads();
  for (int o = 128; o > 0; o >>= 1) {
    if ((int)threadIdx.x < o) s[threadIdx.x] += s[threadIdx.x + o];
    __syncthreads();
  }
  return s[0];
}

// ---------------- one-time-per-call prep kernels ----------------

// zero the 2+2 padded rows of both ping/pong activation buffers
__global__ void pad_zero_kernel(u16* a, u16* b) {
  int tid = blockIdx.x * 256 + threadIdx.x;
  const int per = B_ * 4 * H_;
  if (tid >= 2 * per) return;
  u16* dst = (tid < per) ? a : b;
  int r  = tid % per;
  int bb = r / (4 * H_);
  int q  = r % (4 * H_);
  int rr = q / H_;
  int c  = q % H_;
  int row = (rr < 2) ? rr : (TP_ - 4 + rr);   // 0,1,2050,2051
  dst[((size_t)bb * TP_ + row) * H_ + c] = 0;
}

// pack w_rest (4,H,H,K) fp32 into per-lane WMMA B-operand order, bf16:
// layout [layer][kb][nb][lane][elem], K-dim index kappa = k*320 + ci
__global__ void pack_w_kernel(const float* __restrict__ wrest, u16* __restrict__ wpack) {
  int tid = blockIdx.x * 256 + threadIdx.x;
  if (tid >= 4 * WPER_) return;
  int l = tid / WPER_;
  int r = tid % WPER_;
  int e    = r & 15;
  int lane = (r >> 4) & 31;
  int nb   = (r >> 9) % NBLK_;
  int kb   = (r >> 9) / NBLK_;
  int koff = ((lane >> 4) << 4) + e;          // lanes 0-15: K 0..15, lanes 16-31: K 16..31
  int kap  = kb * 32 + koff;
  int kk   = kap / H_;
  int ci   = kap % H_;
  int co   = nb * 16 + (lane & 15);
  float v = wrest[(((size_t)l * H_ + co) * H_ + ci) * K_ + kk];
  wpack[(size_t)l * WPER_ + r] = f2bf(v);
}

__global__ void pack_emb_kernel(const float* __restrict__ emb, u16* __restrict__ embbf) {
  int tid = blockIdx.x * 256 + threadIdx.x;
  if (tid < NE_ * D_) embbf[tid] = f2bf(emb[tid]);
}

__global__ void embnorm_kernel(const float* __restrict__ emb, float* __restrict__ embn) {
  int e = blockIdx.x;                          // 2048 blocks, 256 threads (== D_)
  float v = emb[(size_t)e * D_ + threadIdx.x];
  float s = blockReduceSum256(v * v);
  if (threadIdx.x == 0) embn[e] = s;
}

// ---------------- conv stack ----------------

// conv0: 1 -> 320, K=5, trivial FLOPs; writes pre-BN fp32 in (B,T,H)
__global__ void conv0_kernel(const float* __restrict__ x, const float* __restrict__ w0,
                             const float* __restrict__ b0, float* __restrict__ pre) {
  size_t p = (size_t)blockIdx.x * 256 + threadIdx.x;
  if (p >= NTOT_) return;
  int c = (int)(p % H_);
  size_t i = p / H_;
  int t = (int)(i % T_);
  int b = (int)(i / T_);
  float s = b0[c];
#pragma unroll
  for (int k = 0; k < K_; ++k) {
    int tt = t + k - 2;
    if ((unsigned)tt < (unsigned)T_) s += w0[c * K_ + k] * x[(size_t)b * T_ + tt];
  }
  pre[p] = s;
}

// 320->320 conv as bf16 WMMA GEMM.  C(t,co) = sum_kappa A(t,kappa)*B(kappa,co)
// A = padded (B,TP,H) bf16 activations (im2col rows are contiguous chunks).
// B = pre-packed weights, staged per K-step into LDS with async DMA (double
// buffered) and shared by all 4 waves.  Wave: 16(t) x 64(co); WG: 64x64.
__global__ void __launch_bounds__(128) conv_wmma_kernel(
    const u16* __restrict__ act, const u16* __restrict__ wpack,
    const float* __restrict__ bias, float* __restrict__ pre) {
  __shared__ u16 bsm[2][4 * 32 * 16];      // 2 x 4KB weight stages
  const int tid  = threadIdx.x;
  const int lane = tid & 31;
  const int wave = tid >> 5;
  const int bz   = blockIdx.z;
  const int tbase = blockIdx.x * 64 + wave * 16;
  const int hi = lane >> 4;       // lane half
  const int lo = lane & 15;

  v8f acc0 = {0.f,0.f,0.f,0.f,0.f,0.f,0.f,0.f};
  v8f acc1 = acc0, acc2 = acc0, acc3 = acc0;

  const unsigned lb[2] = { lds_addr_of(&bsm[0][0]), lds_addr_of(&bsm[1][0]) };
  const unsigned co0 = tid * 32;           // this thread's 32B slice of a 4KB stage

  // prologue: stage 0 <- K-block 0
  {
    const u16* src = wpack + (size_t)(0 * NBLK_ + blockIdx.y * 4) * 512;
    async_load_b128(lb[0] + co0,      src, co0);
    async_load_b128(lb[0] + co0 + 16, src, co0 + 16);
  }
  wait_async0();
  __syncthreads();

  const int mrow = tbase + lo;    // A-matrix row (same for both lane halves)
  for (int kb = 0; kb < KBLK_; ++kb) {
    const int st = kb & 1;
    if (kb + 1 < KBLK_) {         // async-prefetch next weight K-block
      const u16* src = wpack + (size_t)((kb + 1) * NBLK_ + blockIdx.y * 4) * 512;
      async_load_b128(lb[st ^ 1] + co0,      src, co0);
      async_load_b128(lb[st ^ 1] + co0 + 16, src, co0 + 16);
    }
    const int kk  = (kb * 32) / H_;          // filter tap (32 | 320, never splits)
    const int ci0 = (kb * 32) % H_;          // channel base
    // padded row = t + kk  (== (t+kk-2) + 2)
    const u16* arow = act + ((size_t)bz * TP_ + (mrow + kk)) * H_ + ci0 + (hi << 4);
    v16bf A = load_bf16x16(arow, arow + 8);  // K {0..15} / {16..31}

    const u16* bl = &bsm[st][lane * 16];     // per-lane 32B operand from LDS
    v16bf B0 = load_bf16x16(bl,        bl + 8);
    v16bf B1 = load_bf16x16(bl + 512,  bl + 520);
    v16bf B2 = load_bf16x16(bl + 1024, bl + 1032);
    v16bf B3 = load_bf16x16(bl + 1536, bl + 1544);
    acc0 = __builtin_amdgcn_wmma_f32_16x16x32_bf16(false, A, false, B0, (short)0, acc0, false, false);
    acc1 = __builtin_amdgcn_wmma_f32_16x16x32_bf16(false, A, false, B1, (short)0, acc1, false, false);
    acc2 = __builtin_amdgcn_wmma_f32_16x16x32_bf16(false, A, false, B2, (short)0, acc2, false, false);
    acc3 = __builtin_amdgcn_wmma_f32_16x16x32_bf16(false, A, false, B3, (short)0, acc3, false, false);

    wait_async0();                // next stage landed; barrier before reuse
    __syncthreads();
  }

  const float* bp = bias + blockIdx.y * 64;
  const float bv0 = bp[lo], bv1 = bp[16 + lo], bv2 = bp[32 + lo], bv3 = bp[48 + lo];
#pragma unroll
  for (int j = 0; j < 8; ++j) {                 // C layout: VGPR j -> M = j + 8*hi
    const int t = tbase + j + (hi << 3);
    const size_t rb = ((size_t)bz * T_ + t) * H_ + blockIdx.y * 64 + lo;
    pre[rb]      = acc0[j] + bv0;
    pre[rb + 16] = acc1[j] + bv1;
    pre[rb + 32] = acc2[j] + bv2;
    pre[rb + 48] = acc3[j] + bv3;
  }
}

// BatchNorm training stats per channel -> fused scale/shift
__global__ void __launch_bounds__(256) bn_stats_kernel(
    const float* __restrict__ pre, const float* __restrict__ gamma,
    const float* __restrict__ beta, float* __restrict__ ss) {
  const int c = blockIdx.x;
  float s1 = 0.f, s2 = 0.f;
  for (int i = threadIdx.x; i < B_ * T_; i += 256) {
    float v = pre[(size_t)i * H_ + c];
    s1 += v; s2 += v * v;
  }
  s1 = blockReduceSum256(s1);
  s2 = blockReduceSum256(s2);
  if (threadIdx.x == 0) {
    const float inv = 1.0f / (float)(B_ * T_);
    float m   = s1 * inv;
    float var = s2 * inv - m * m;              // biased var (training stats)
    float sc  = gamma[c] * rsqrtf(var + 1e-5f);
    ss[c]      = sc;
    ss[H_ + c] = beta[c] - m * sc;
  }
}

// apply BN + ReLU; write bf16 into padded (B,TP,H); optionally write fp32 back (last layer)
__global__ void bn_apply_kernel(float* __restrict__ pre, const float* __restrict__ ss,
                                u16* __restrict__ actout, int writeback) {
  size_t p = (size_t)blockIdx.x * 256 + threadIdx.x;
  if (p >= NTOT_) return;
  int c = (int)(p % H_);
  size_t i = p / H_;
  int t = (int)(i % T_);
  int b = (int)(i / T_);
  float v = pre[p] * ss[c] + ss[H_ + c];
  v = v > 0.f ? v : 0.f;
  actout[((size_t)b * TP_ + t + 2) * H_ + c] = f2bf(v);
  if (writeback) pre[p] = v;
}

// ---------------- VQ ----------------

// argmin_e ( ||emb_e||^2 - 2 * flat_n . emb_e ) via bf16 WMMA over K=256.
// Each wave owns 16 flat rows; the 16-row codebook chunk (8KB) for each e-step
// is staged into LDS with async DMA (double buffered), shared by all 4 waves.
__global__ void __launch_bounds__(128) vq_argmin_kernel(
    const u16* __restrict__ act, const u16* __restrict__ embbf,
    const float* __restrict__ embn, int* __restrict__ idx) {
  __shared__ u16   esm[2][16 * D_];   // 2 x 8KB codebook stages
  __shared__ float smin[128][8];
  __shared__ int   sidx[128][8];
  const int tid  = threadIdx.x;
  const int lane = tid & 31;
  const int wave = tid >> 5;
  const int hi = lane >> 4, lo = lane & 15;
  const int nb = blockIdx.x * 64 + wave * 16;
  const int n  = nb + lo;
  const int b  = n / NPB_;
  const size_t abase = (size_t)b * (TP_ * H_) + 2 * H_ + (size_t)(n - b * NPB_) * D_;

  const unsigned lb[2] = { lds_addr_of(&esm[0][0]), lds_addr_of(&esm[1][0]) };
  const unsigned co0 = tid * 64;        // this thread's 64B slice of an 8KB stage

  float rmin[8]; int ridx[8];
#pragma unroll
  for (int j = 0; j < 8; ++j) { rmin[j] = 3.4e38f; ridx[j] = 0; }

  // prologue: stage 0 <- e-chunk 0
  {
    const u16* src = embbf;             // rows [0,16) x 256
#pragma unroll
    for (int i = 0; i < 4; ++i) async_load_b128(lb[0] + co0 + i * 16, src, co0 + i * 16);
  }
  wait_async0();
  __syncthreads();

  for (int ec = 0; ec < NE_ / 16; ++ec) {
    const int st = ec & 1;
    if (ec + 1 < NE_ / 16) {            // async-prefetch next 16 codebook rows
      const u16* src = embbf + (size_t)(ec + 1) * 16 * D_;
#pragma unroll
      for (int i = 0; i < 4; ++i) async_load_b128(lb[st ^ 1] + co0 + i * 16, src, co0 + i * 16);
    }
    const int ebase = ec * 16;
    v8f acc = {0.f,0.f,0.f,0.f,0.f,0.f,0.f,0.f};
#pragma unroll
    for (int ks = 0; ks < D_ / 32; ++ks) {
      const int d0 = ks * 32;
      const u16* pa = act + abase + d0 + (hi << 4);
      v16bf A = load_bf16x16(pa, pa + 8);
      const u16* pb = &esm[st][lo * D_ + d0 + (hi << 4)];
      v16bf Bv = load_bf16x16(pb, pb + 8);
      acc = __builtin_amdgcn_wmma_f32_16x16x32_bf16(false, A, false, Bv, (short)0, acc, false, false);
    }
    const float en = embn[ebase + lo];
    const int   e  = ebase + lo;
#pragma unroll
    for (int j = 0; j < 8; ++j) {
      float v = en - 2.0f * acc[j];
      if (v < rmin[j]) { rmin[j] = v; ridx[j] = e; }   // ascending e => first-min kept
    }
    wait_async0();
    __syncthreads();
  }
#pragma unroll
  for (int j = 0; j < 8; ++j) { smin[tid][j] = rmin[j]; sidx[tid][j] = ridx[j]; }
  __syncthreads();
  if (lane < 16) {                 // row r = lane of this wave's 16 rows
    const int r = lane, s = r >> 3, j = r & 7;
    const int bt = wave * 32 + s * 16;
    float best = 3.5e38f; int bi = NE_;
    for (int q = 0; q < 16; ++q) {
      float v = smin[bt + q][j]; int e2 = sidx[bt + q][j];
      if (v < best || (v == best && e2 < bi)) { best = v; bi = e2; }
    }
    idx[nb + r] = bi;
  }
}

__global__ void zero_small_kernel(int* hist, float* lacc) {
  int tid = blockIdx.x * 256 + threadIdx.x;
  if (tid < NE_) hist[tid] = 0;
  if (tid == NE_) *lacc = 0.f;
}

// gather quantized output and accumulate sum((q-inp)^2).
// Output-major mapping: stores to (B,H,T) are fully coalesced; the strided
// gather reads of inp (42MB) stay resident in the 192MB L2.
__global__ void __launch_bounds__(256) quant_loss_kernel(
    const int* __restrict__ idx, const float* __restrict__ emb,
    const float* __restrict__ inp, float* __restrict__ outq, float* __restrict__ lacc) {
  size_t o = (size_t)blockIdx.x * 256 + threadIdx.x;   // index into (B,H,T)
  float val = 0.f;
  if (o < NTOT_) {
    int t = (int)(o % T_);
    size_t j = o / T_;
    int c = (int)(j % H_);
    int b = (int)(j / H_);
    size_t p = ((size_t)b * T_ + t) * H_ + c;          // (B,T,H) flat position
    int n = (int)(p >> 8), d = (int)(p & 255);
    float q = emb[(size_t)idx[n] * D_ + d];
    float diff = q - inp[p];
    val = diff * diff;
    outq[o] = q;                                       // quant_st == quant in value
  }
  float s = blockReduceSum256(val);
  if (threadIdx.x == 0) atomicAdd(lacc, s);
}

__global__ void hist_kernel(const int* __restrict__ idx, int* __restrict__ hist) {
  int tid = blockIdx.x * 256 + threadIdx.x;
  if (tid < NFLAT_) atomicAdd(&hist[idx[tid]], 1);
}

__global__ void finalize_kernel(const float* __restrict__ lacc, const int* __restrict__ hist,
                                float* __restrict__ out_loss, float* __restrict__ out_ppl) {
  float part = 0.f;
  for (int e = threadIdx.x; e < NE_; e += 256) {
    float avg = (float)hist[e] * (1.0f / (float)NFLAT_);
    part += avg * logf(avg + 1e-10f);
  }
  float s = blockReduceSum256(part);
  if (threadIdx.x == 0) {
    *out_ppl  = expf(-s);
    *out_loss = 1.25f * lacc[0] / (float)NTOT_;   // q_latent + 0.25*e_latent, equal values
  }
}

// ---------------- launcher ----------------
extern "C" void kernel_launch(void* const* d_in, const int* in_sizes, int n_in,
                              void* d_out, int out_size, void* d_ws, size_t ws_size,
                              hipStream_t stream) {
  (void)in_sizes; (void)n_in; (void)ws_size;
  const float* x      = (const float*)d_in[0];
  const float* w0     = (const float*)d_in[1];
  const float* b0     = (const float*)d_in[2];
  const float* w_rest = (const float*)d_in[3];
  const float* b_rest = (const float*)d_in[4];
  const float* gamma  = (const float*)d_in[5];
  const float* beta   = (const float*)d_in[6];
  const float* emb    = (const float*)d_in[7];

  float* out      = (float*)d_out;
  float* out_loss = out;
  float* outq     = out + 1;
  float* out_ppl  = out + (size_t)out_size - 1;

  // workspace carve-up (256B aligned)
  char* w = (char*)d_ws;
  size_t off = 0;
  auto carve = [&](size_t bytes) { char* p = w + off; off = (off + bytes + 255) & ~(size_t)255; return p; };
  u16*   actA  = (u16*)  carve((size_t)B_ * TP_ * H_ * sizeof(u16));
  u16*   actB  = (u16*)  carve((size_t)B_ * TP_ * H_ * sizeof(u16));
  float* pre   = (float*)carve(NTOT_ * sizeof(float));
  u16*   wpack = (u16*)  carve((size_t)4 * WPER_ * sizeof(u16));
  u16*   embbf = (u16*)  carve((size_t)NE_ * D_ * sizeof(u16));
  float* embn  = (float*)carve((size_t)NE_ * sizeof(float));
  float* ss    = (float*)carve((size_t)2 * H_ * sizeof(float));
  int*   idxb  = (int*)  carve((size_t)NFLAT_ * sizeof(int));
  int*   hist  = (int*)  carve((size_t)NE_ * sizeof(int));
  float* lacc  = (float*)carve(256);

  const int nblk_elem = (int)((NTOT_ + 255) / 256);   // 40960

  // prep
  pad_zero_kernel<<<(2 * B_ * 4 * H_ + 255) / 256, 256, 0, stream>>>(actA, actB);
  pack_w_kernel<<<(4 * WPER_ + 255) / 256, 256, 0, stream>>>(w_rest, wpack);
  pack_emb_kernel<<<(NE_ * D_ + 255) / 256, 256, 0, stream>>>(emb, embbf);
  embnorm_kernel<<<NE_, 256, 0, stream>>>(emb, embn);

  // layer 0 (1->320)
  conv0_kernel<<<nblk_elem, 256, 0, stream>>>(x, w0, b0, pre);
  bn_stats_kernel<<<H_, 256, 0, stream>>>(pre, gamma, beta, ss);
  bn_apply_kernel<<<nblk_elem, 256, 0, stream>>>(pre, ss, actA, 0);

  // layers 1..4 (320->320 WMMA)
  u16* bufs[2] = {actA, actB};
  int cur = 0;
  dim3 cgrid(T_ / 64, H_ / 64, B_);
  for (int l = 0; l < 4; ++l) {
    conv_wmma_kernel<<<cgrid, 128, 0, stream>>>(bufs[cur], wpack + (size_t)l * WPER_,
                                                b_rest + (size_t)l * H_, pre);
    bn_stats_kernel<<<H_, 256, 0, stream>>>(pre, gamma + (size_t)(l + 1) * H_,
                                            beta + (size_t)(l + 1) * H_, ss);
    bn_apply_kernel<<<nblk_elem, 256, 0, stream>>>(pre, ss, bufs[1 - cur], (l == 3) ? 1 : 0);
    cur = 1 - cur;
  }
  // final bf16 activations in bufs[0] (== actA); fp32 inp in `pre`

  // VQ
  vq_argmin_kernel<<<NFLAT_ / 64, 128, 0, stream>>>(actA, embbf, embn, idxb);
  zero_small_kernel<<<(NE_ + 1 + 255) / 256, 256, 0, stream>>>(hist, lacc);
  quant_loss_kernel<<<nblk_elem, 256, 0, stream>>>(idxb, emb, pre, outq, lacc);
  hist_kernel<<<(NFLAT_ + 255) / 256, 256, 0, stream>>>(idxb, hist);
  finalize_kernel<<<1, 256, 0, stream>>>(lacc, hist, out_loss, out_ppl);
}
